// MambaWithFFN_3547642986682
// MI455X (gfx1250) — matrix operand order repaired and checked by reading, not compiled, not run on hardware
//
#include <hip/hip_runtime.h>
#include <hip/hip_bf16.h>
#include <math.h>

typedef __attribute__((ext_vector_type(16))) __bf16 v16bf;
typedef __attribute__((ext_vector_type(8)))  float  v8f;
typedef __attribute__((ext_vector_type(8)))  unsigned int v8u;
typedef __attribute__((ext_vector_type(4)))  unsigned int v4u;

#define DMODEL 1024
#define DINNER 2048
#define DSTATE 16
#define DCONV  4
#define DTRANK 64
#define DFF    2048
#define BB     4
#define LL     2048
#define MROWS  (BB*LL)   // 8192 rows
#define DBCW   (DTRANK + 2*DSTATE)  // 96

// ---------- helpers ----------
__device__ __forceinline__ unsigned short f2bf(float f) {
    union { float f; unsigned int u; } c; c.f = f;
    unsigned int u = c.u;
    unsigned int r = (u + 0x7FFFu + ((u >> 16) & 1u));
    if ((u & 0x7F800000u) == 0x7F800000u) r = u;   // keep inf/nan exponent
    return (unsigned short)(r >> 16);
}

template <int ACT>
__device__ __forceinline__ float apply_act(float v) {
    if (ACT == 1)      v = (v > 20.f) ? v : log1pf(__expf(v));            // softplus
    else if (ACT == 2) v = 0.5f * v * (1.f + erff(v * 0.70710678118f));   // exact gelu
    return v;
}

// ---------- fp32 -> bf16 elementwise ----------
__global__ void cvt_f32_to_bf16_kernel(const float* __restrict__ src,
                                       unsigned short* __restrict__ dst, int n) {
    int i = blockIdx.x * blockDim.x + threadIdx.x;
    if (i < n) dst[i] = f2bf(src[i]);
}

// ---------- layer norm (row = 1024), writes bf16 ----------
__global__ void __launch_bounds__(256) ln_bf16_kernel(
        const float* __restrict__ x,
        const float* __restrict__ w,
        const float* __restrict__ b,
        unsigned short* __restrict__ out) {
    __shared__ float red[256];
    const int row = blockIdx.x;
    const float* xr = x + (size_t)row * DMODEL;
    float s = 0.f;
    for (int i = threadIdx.x; i < DMODEL; i += 256) s += xr[i];
    red[threadIdx.x] = s; __syncthreads();
    for (int st = 128; st > 0; st >>= 1) {
        if (threadIdx.x < st) red[threadIdx.x] += red[threadIdx.x + st];
        __syncthreads();
    }
    float mu = red[0] * (1.f / DMODEL);
    __syncthreads();
    float v = 0.f;
    for (int i = threadIdx.x; i < DMODEL; i += 256) { float d = xr[i] - mu; v += d * d; }
    red[threadIdx.x] = v; __syncthreads();
    for (int st = 128; st > 0; st >>= 1) {
        if (threadIdx.x < st) red[threadIdx.x] += red[threadIdx.x + st];
        __syncthreads();
    }
    float rstd = rsqrtf(red[0] * (1.f / DMODEL) + 1e-5f);
    for (int i = threadIdx.x; i < DMODEL; i += 256) {
        float y = (xr[i] - mu) * rstd * w[i] + b[i];
        out[(size_t)row * DMODEL + i] = f2bf(y);
    }
}

// =======================================================================
// Blocked WMMA GEMM: each wave computes a 32M x 64N macro-tile.
// out[M,N] = act(A[M,K] @ W[N,K]^T + bias) + resid
// LDA/LDB/N/K compile-time: all 8 fragment loads per K-step use immediate
// offsets from 2 base pointers -> no address spills, deep pipelining.
// Requires M%32==0, N%64==0, K%32==0.
// =======================================================================
template <int LDA, int LDB, int N, int K, int ACT>
__global__ void __launch_bounds__(256) wmma_gemm_bf16_blk_kernel(
    const unsigned short* __restrict__ A,
    const unsigned short* __restrict__ W,
    int M,
    const float* __restrict__ bias,
    const float* __restrict__ resid,
    float* __restrict__ outf,
    unsigned short* __restrict__ outbf) {
    const int lane = threadIdx.x & 31;
    const int wave = threadIdx.x >> 5;
    const int tilesN = N >> 6;                    // 64-wide N tiles
    const int tile = blockIdx.x * 8 + wave;
    if (tile >= (M >> 5) * tilesN) return;        // wave-uniform: EXEC all-ones
    const int mt = tile / tilesN;
    const int nt = tile % tilesN;
    const int half = lane >> 4;                   // K half-select (ISA 7.12.2)
    const int l16  = lane & 15;

    // Two base pointers; everything else is a compile-time immediate offset.
    const unsigned short* abase = A + (size_t)(mt * 32 + l16) * LDA + half * 16;
    const unsigned short* bbase = W + (size_t)(nt * 64 + l16) * LDB + half * 16;

    union FragA { v4u h[2]; v16bf bf; };
    union FragB { v8u  u;   v16bf bf; };

    v8f acc[8];
    #pragma unroll
    for (int i = 0; i < 8; ++i) acc[i] = (v8f){};

    for (int k0 = 0; k0 < K; k0 += 32) {
        FragA a[2]; FragB b[4];
        // A lane (M = l16 [+16]): K = half*16..+7 in VGPR0-3, half*16+8..+15 in VGPR4-7
        #pragma unroll
        for (int mi = 0; mi < 2; ++mi) {
            a[mi].h[0] = *(const v4u*)(abase + mi * (16 * LDA) + k0);
            a[mi].h[1] = *(const v4u*)(abase + mi * (16 * LDA) + k0 + 8);
        }
        // B lane (N = l16 + ni*16): 16 contiguous K values starting at half*16
        #pragma unroll
        for (int ni = 0; ni < 4; ++ni)
            b[ni].u = *(const v8u*)(bbase + ni * (16 * LDB) + k0);
        // prefetch only the streaming A rows (weights are L2-resident)
        __builtin_prefetch(abase + k0 + 128, 0, 0);
        __builtin_prefetch(abase + 16 * LDA + k0 + 128, 0, 0);
        #pragma unroll
        for (int mi = 0; mi < 2; ++mi)
            #pragma unroll
            for (int ni = 0; ni < 4; ++ni)
                acc[mi * 4 + ni] = __builtin_amdgcn_wmma_f32_16x16x32_bf16(
                    false, a[mi].bf, false, b[ni].bf, (short)0,
                    acc[mi * 4 + ni], false, false);
    }

    #pragma unroll
    for (int mi = 0; mi < 2; ++mi) {
        #pragma unroll
        for (int ni = 0; ni < 4; ++ni) {
            const int col = nt * 64 + ni * 16 + l16;
            const float bv = bias ? bias[col] : 0.f;
            const v8f d = acc[mi * 4 + ni];
            #pragma unroll
            for (int j = 0; j < 8; ++j) {
                const int row = mt * 32 + mi * 16 + half * 8 + j;  // VGPR j -> M=half*8+j
                float v = apply_act<ACT>(d[j] + bv);
                const size_t o = (size_t)row * N + col;
                if (resid) v += resid[o];
                if (outf)  outf[o]  = v;
                if (outbf) outbf[o] = f2bf(v);
            }
        }
    }
}

// ---------- 16x16-per-wave WMMA GEMM for N not divisible by 64 (x_proj) ----------
template <int LDA, int LDB, int N, int K, int ACT>
__global__ void __launch_bounds__(256) wmma_gemm_bf16_small_kernel(
    const unsigned short* __restrict__ A,
    const unsigned short* __restrict__ W,
    int M,
    const float* __restrict__ bias,
    const float* __restrict__ resid,
    float* __restrict__ outf,
    unsigned short* __restrict__ outbf) {
    const int lane = threadIdx.x & 31;
    const int wave = threadIdx.x >> 5;
    const int tilesN = N >> 4;
    const int tile = blockIdx.x * 8 + wave;
    if (tile >= (M >> 4) * tilesN) return;
    const int mt = tile / tilesN;
    const int nt = tile % tilesN;
    const int half = lane >> 4;
    const int l16  = lane & 15;

    const unsigned short* abase = A + (size_t)(mt * 16 + l16) * LDA + half * 16;
    const unsigned short* bbase = W + (size_t)(nt * 16 + l16) * LDB + half * 16;

    union FragA { v4u h[2]; v16bf bf; };
    union FragB { v8u  u;   v16bf bf; };

    v8f acc = {};
    for (int k0 = 0; k0 < K; k0 += 32) {
        FragA a; FragB b;
        a.h[0] = *(const v4u*)(abase + k0);
        a.h[1] = *(const v4u*)(abase + k0 + 8);
        b.u    = *(const v8u*)(bbase + k0);
        __builtin_prefetch(abase + k0 + 128, 0, 0);
        acc = __builtin_amdgcn_wmma_f32_16x16x32_bf16(
                  false, a.bf, false, b.bf, (short)0, acc, false, false);
    }

    const int col = nt * 16 + l16;
    const float bv = bias ? bias[col] : 0.f;
    #pragma unroll
    for (int j = 0; j < 8; ++j) {
        const int row = mt * 16 + half * 8 + j;
        float v = apply_act<ACT>(acc[j] + bv);
        const size_t o = (size_t)row * N + col;
        if (resid) v += resid[o];
        if (outf)  outf[o]  = v;
        if (outbf) outbf[o] = f2bf(v);
    }
}

// ---------- depthwise causal conv (k=4) + SiLU ----------
__global__ void __launch_bounds__(256) conv_silu_kernel(
        const float* __restrict__ xz,
        const float* __restrict__ cw,
        const float* __restrict__ cb,
        float* __restrict__ xcf,
        unsigned short* __restrict__ xcb) {
    size_t i = (size_t)blockIdx.x * blockDim.x + threadIdx.x;
    if (i >= (size_t)MROWS * DINNER) return;
    const int d  = (int)(i % DINNER);
    const size_t bl = i / DINNER;
    const int l  = (int)(bl % LL);
    const size_t brow = bl - (size_t)l;           // b*L
    float acc = cb[d];
    #pragma unroll
    for (int k = 0; k < DCONV; ++k) {
        const int ls = l + k - (DCONV - 1);
        if (ls >= 0)
            acc += xz[(brow + (size_t)ls) * (size_t)(2 * DINNER) + d] * cw[d * DCONV + k];
    }
    const float s = acc / (1.f + __expf(-acc));
    xcf[i] = s;
    xcb[i] = f2bf(s);
}

// ---------- slice dbc[:, :64] -> bf16 ----------
__global__ void slice_dtraw_kernel(const float* __restrict__ dbc,
                                   unsigned short* __restrict__ dtraw) {
    int i = blockIdx.x * blockDim.x + threadIdx.x;
    if (i >= MROWS * DTRANK) return;
    const int r = i / DTRANK, c = i % DTRANK;
    dtraw[i] = f2bf(dbc[(size_t)r * DBCW + c]);
}

// ---------- selective scan: one channel (b,d) per thread ----------
__global__ void __launch_bounds__(256) scan_kernel(
        const float* __restrict__ dt,
        const float* __restrict__ xcf,
        const float* __restrict__ dbc,
        const float* __restrict__ xz,
        const float* __restrict__ A_log,
        const float* __restrict__ Dp,
        unsigned short* __restrict__ yg) {
    __shared__ float bcm[32][33];                  // B(16)+C(16) per step, 32 steps
    const int b     = blockIdx.x / (DINNER / 256);
    const int chunk = blockIdx.x % (DINNER / 256);
    const int d     = chunk * 256 + threadIdx.x;
    float Aa[DSTATE], h[DSTATE];
    #pragma unroll
    for (int n = 0; n < DSTATE; ++n) { Aa[n] = -__expf(A_log[d * DSTATE + n]); h[n] = 0.f; }
    const float dpd = Dp[d];
    for (int l0 = 0; l0 < LL; l0 += 32) {
        __syncthreads();
        #pragma unroll
        for (int r = 0; r < 4; ++r) {
            const int e = threadIdx.x + r * 256;
            const int i = e >> 5, j = e & 31;
            bcm[i][j] = dbc[((size_t)(b * LL + l0 + i)) * DBCW + DTRANK + j];
        }
        __syncthreads();
        for (int i = 0; i < 32; ++i) {
            const size_t rowbl = (size_t)(b * LL + l0 + i);
            const float dtv = dt[rowbl * DINNER + d];
            const float xcv = xcf[rowbl * DINNER + d];
            float y = 0.f;
            #pragma unroll
            for (int n = 0; n < DSTATE; ++n) {
                const float dA = __expf(dtv * Aa[n]);
                h[n] = dA * h[n] + (dtv * bcm[i][n]) * xcv;
                y += h[n] * bcm[i][16 + n];
            }
            y += xcv * dpd;
            const float zv = xz[rowbl * (size_t)(2 * DINNER) + DINNER + d];
            y *= zv / (1.f + __expf(-zv));
            yg[rowbl * DINNER + d] = f2bf(y);
        }
    }
}

// ---------- host launcher ----------
extern "C" void kernel_launch(void* const* d_in, const int* in_sizes, int n_in,
                              void* d_out, int out_size, void* d_ws, size_t ws_size,
                              hipStream_t stream) {
    const float* x         = (const float*)d_in[0];
    const float* ln1_w     = (const float*)d_in[1];
    const float* ln1_b     = (const float*)d_in[2];
    const float* in_proj_w = (const float*)d_in[3];
    const float* conv_w    = (const float*)d_in[4];
    const float* conv_b    = (const float*)d_in[5];
    const float* x_proj_w  = (const float*)d_in[6];
    const float* dt_proj_w = (const float*)d_in[7];
    const float* dt_proj_b = (const float*)d_in[8];
    const float* A_log     = (const float*)d_in[9];
    const float* Dvec      = (const float*)d_in[10];
    const float* out_proj_w= (const float*)d_in[11];
    const float* ln2_w     = (const float*)d_in[12];
    const float* ln2_b     = (const float*)d_in[13];
    const float* ffn_w1    = (const float*)d_in[14];
    const float* ffn_b1    = (const float*)d_in[15];
    const float* ffn_w2    = (const float*)d_in[16];
    const float* ffn_b2    = (const float*)d_in[17];

    char* ws = (char*)d_ws;
    size_t off = 0;
    auto alloc = [&](size_t bytes) -> void* {
        size_t r = (off + 255) & ~(size_t)255;
        off = r + bytes;
        return (void*)(ws + r);
    };
    unsigned short* w_in_bf   = (unsigned short*)alloc(2ull * (2*DINNER) * DMODEL);
    unsigned short* w_xp_bf   = (unsigned short*)alloc(2ull * DBCW * DINNER);
    unsigned short* w_dt_bf   = (unsigned short*)alloc(2ull * DINNER * DTRANK);
    unsigned short* w_op_bf   = (unsigned short*)alloc(2ull * DMODEL * DINNER);
    unsigned short* w_f1_bf   = (unsigned short*)alloc(2ull * DFF * DMODEL);
    unsigned short* w_f2_bf   = (unsigned short*)alloc(2ull * DMODEL * DFF);
    unsigned short* xn_bf     = (unsigned short*)alloc(2ull * MROWS * DMODEL);
    float*          xz_f      = (float*)         alloc(4ull * MROWS * 2 * DINNER);
    float*          xc_f      = (float*)         alloc(4ull * MROWS * DINNER);
    unsigned short* xc_bf     = (unsigned short*)alloc(2ull * MROWS * DINNER);
    float*          dbc_f     = (float*)         alloc(4ull * MROWS * DBCW);
    unsigned short* dtraw_bf  = (unsigned short*)alloc(2ull * MROWS * DTRANK);
    float*          dt_f      = (float*)         alloc(4ull * MROWS * DINNER);
    unsigned short* yg_bf     = (unsigned short*)alloc(2ull * MROWS * DINNER);
    float*          x2_f      = (float*)         alloc(4ull * MROWS * DMODEL);
    unsigned short* h2_bf     = (unsigned short*)alloc(2ull * MROWS * DMODEL);
    unsigned short* g_bf      = (unsigned short*)alloc(2ull * MROWS * DFF);

    auto cvt = [&](const float* s, unsigned short* d, int n) {
        cvt_f32_to_bf16_kernel<<<(n + 255) / 256, 256, 0, stream>>>(s, d, n);
    };
    cvt(in_proj_w,  w_in_bf, 2 * DINNER * DMODEL);
    cvt(x_proj_w,   w_xp_bf, DBCW * DINNER);
    cvt(dt_proj_w,  w_dt_bf, DINNER * DTRANK);
    cvt(out_proj_w, w_op_bf, DMODEL * DINNER);
    cvt(ffn_w1,     w_f1_bf, DFF * DMODEL);
    cvt(ffn_w2,     w_f2_bf, DMODEL * DFF);

    const int nblk1024 = ((MROWS / 32) * (1024 / 64) + 7) / 8;  // N=1024
    const int nblk2048 = ((MROWS / 32) * (2048 / 64) + 7) / 8;  // N=2048
    const int nblk4096 = ((MROWS / 32) * (4096 / 64) + 7) / 8;  // N=4096

    // 1) LN1 -> bf16
    ln_bf16_kernel<<<MROWS, 256, 0, stream>>>(x, ln1_w, ln1_b, xn_bf);
    // 2) in_proj: xz = xn @ Win^T   (8192 x 4096, K=1024)
    wmma_gemm_bf16_blk_kernel<DMODEL, DMODEL, 2*DINNER, DMODEL, 0>
        <<<nblk4096, 256, 0, stream>>>(xn_bf, w_in_bf, MROWS,
                                       nullptr, nullptr, xz_f, nullptr);
    // 3) causal conv + SiLU
    {
        const size_t nthreads = (size_t)MROWS * DINNER;
        conv_silu_kernel<<<(unsigned)((nthreads + 255) / 256), 256, 0, stream>>>(
            xz_f, conv_w, conv_b, xc_f, xc_bf);
    }
    // 4) x_proj: dbc = xc @ Wxp^T   (8192 x 96, K=2048)  -- N=96 -> 16-wide kernel
    {
        const int tiles = (MROWS / 16) * (DBCW / 16);
        wmma_gemm_bf16_small_kernel<DINNER, DINNER, DBCW, DINNER, 0>
            <<<(tiles + 7) / 8, 256, 0, stream>>>(xc_bf, w_xp_bf, MROWS,
                                                  nullptr, nullptr, dbc_f, nullptr);
    }
    // 5) slice dt_raw -> bf16
    slice_dtraw_kernel<<<(MROWS * DTRANK + 255) / 256, 256, 0, stream>>>(dbc_f, dtraw_bf);
    // 6) dt = softplus(dt_raw @ Wdt^T + b)   (8192 x 2048, K=64)
    wmma_gemm_bf16_blk_kernel<DTRANK, DTRANK, DINNER, DTRANK, 1>
        <<<nblk2048, 256, 0, stream>>>(dtraw_bf, w_dt_bf, MROWS,
                                       dt_proj_b, nullptr, dt_f, nullptr);
    // 7) selective scan + D-skip + SiLU(z) gate -> yg (bf16)
    scan_kernel<<<BB * (DINNER / 256), 256, 0, stream>>>(
        dt_f, xc_f, dbc_f, xz_f, A_log, Dvec, yg_bf);
    // 8) out_proj + residual(x) -> x2   (8192 x 1024, K=2048)
    wmma_gemm_bf16_blk_kernel<DINNER, DINNER, DMODEL, DINNER, 0>
        <<<nblk1024, 256, 0, stream>>>(yg_bf, w_op_bf, MROWS,
                                       nullptr, x, x2_f, nullptr);
    // 9) LN2 -> bf16
    ln_bf16_kernel<<<MROWS, 256, 0, stream>>>(x2_f, ln2_w, ln2_b, h2_bf);
    // 10) FFN up + GELU -> g (bf16)   (8192 x 2048, K=1024)
    wmma_gemm_bf16_blk_kernel<DMODEL, DMODEL, DFF, DMODEL, 2>
        <<<nblk2048, 256, 0, stream>>>(h2_bf, w_f1_bf, MROWS,
                                       ffn_b1, nullptr, nullptr, g_bf);
    // 11) FFN down + bias + residual(x2) -> out   (8192 x 1024, K=2048)
    wmma_gemm_bf16_blk_kernel<DFF, DFF, DMODEL, DFF, 0>
        <<<nblk1024, 256, 0, stream>>>(g_bf, w_f2_bf, MROWS,
                                       ffn_b2, x2_f, (float*)d_out, nullptr);
}
